// LePEAttentionBlock_6107443495140
// MI455X (gfx1250) — matrix operand order repaired and checked by reading, not compiled, not run on hardware
//
#include <hip/hip_runtime.h>

typedef unsigned short ushort_t;
typedef unsigned long long u64_t;
typedef __attribute__((ext_vector_type(16))) __bf16 v16bf;
typedef __attribute__((ext_vector_type(8)))  float  v8f;

#define NHEAD   8
#define HD      32
#define LWIN    784          // tokens per window (7*112)
#define LPAD    896          // padded to 7*128
#define NWIN    32           // B * n_strips
#define NHW     (NWIN*NHEAD) // 256 head-windows
#define MTOK    25088        // total tokens = 32*784
#define CDIM    256
#define QKVN    768
#define SCALE   0.17677669529663687f  // 32^-0.5

struct alignas(32) U32B { uint4 a, b; };

static __device__ inline v16bf mk16(uint4 a, uint4 b) {
    U32B t{a, b};
    return __builtin_bit_cast(v16bf, t);
}

static __device__ inline v8f wmma_bf16(v16bf a, v16bf b, v8f c) {
    // D = A(16x32 bf16) * B(32x16 bf16) + C(16x16 f32)
    return __builtin_amdgcn_wmma_f32_16x16x32_bf16(false, a, false, b, (short)0, c, false, false);
}

static __device__ inline ushort_t f2bf(float f) {
    unsigned u = __float_as_uint(f);
    unsigned r = u + 0x7FFFu + ((u >> 16) & 1u);  // round-to-nearest-even
    return (ushort_t)(r >> 16);
}

// generic LDS pointer -> 32-bit LDS offset (low 32 bits of the shared aperture addr)
static __device__ inline unsigned lds_off(const void* p) {
    return (unsigned)(size_t)p;
}

// 32B async copy global->LDS per lane (2 x b128, INST_OFFSET applies to both sides)
static __device__ inline void async_ld2(unsigned lds, const void* g) {
    u64_t ga = (u64_t)(size_t)g;
    asm volatile("global_load_async_to_lds_b128 %0, %1, off\n\t"
                 "global_load_async_to_lds_b128 %0, %1, off offset:16"
                 :: "v"(lds), "v"(ga) : "memory");
}
// async loads complete in order: keep the 4 just-issued ops in flight
#define WAIT_ASYNC_PIPE() asm volatile("s_wait_asynccnt 4" ::: "memory")
#define WAIT_ASYNC_ALL()  asm volatile("s_wait_asynccnt 0" ::: "memory")

// ---------------------------------------------------------------- convert
__global__ void cvt_f32_bf16(const float* __restrict__ src, ushort_t* __restrict__ dst, int n) {
    int i = blockIdx.x * blockDim.x + threadIdx.x;
    int stride = gridDim.x * blockDim.x;
    for (; i < n; i += stride) dst[i] = f2bf(src[i]);
}

// ---------------------------------------------------------------- QKV GEMM
// y[m, d] = sum_k xb[m,k] * wqkv[d,k]  (d = which*256 + head*32 + hd)
// scatter: q -> Qs (scaled, [hw][LPAD][32]), k -> Kb ([hw][LPAD][32]), v -> Vtb ([hw][32][LPAD])
__global__ void __launch_bounds__(256) qkv_gemm(
    const ushort_t* __restrict__ xb, const ushort_t* __restrict__ wqkv,
    ushort_t* __restrict__ Qs, ushort_t* __restrict__ Kb, ushort_t* __restrict__ Vtb)
{
    __shared__ ushort_t As[2][128 * 32];   // double-buffered 8KB tiles
    __shared__ ushort_t Bs[2][128 * 32];
    const int m0 = blockIdx.x * 128, n0 = blockIdx.y * 128;
    const int t = threadIdx.x, lane = t & 31, wave = t >> 5;
    const int ln = lane & 15, sel = lane >> 4;
    const int wm = (wave & 1) * 64, wn = (wave >> 1) * 32;

    v8f acc[4][2];
#pragma unroll
    for (int a = 0; a < 4; ++a)
#pragma unroll
        for (int b = 0; b < 2; ++b) acc[a][b] = (v8f){0,0,0,0,0,0,0,0};

    const int lrow = t >> 1, loff = (t & 1) * 16;  // 32B per thread per tile
    const ushort_t* gA = xb + (size_t)(m0 + lrow) * CDIM + loff;
    const ushort_t* gB = wqkv + (size_t)(n0 + lrow) * CDIM + loff;
    const unsigned sA[2] = { lds_off(&As[0][lrow * 32 + loff]), lds_off(&As[1][lrow * 32 + loff]) };
    const unsigned sB[2] = { lds_off(&Bs[0][lrow * 32 + loff]), lds_off(&Bs[1][lrow * 32 + loff]) };

    // prologue: fill buffer 0 asynchronously
    async_ld2(sA[0], gA);
    async_ld2(sB[0], gB);

    for (int k0 = 0; k0 < CDIM; k0 += 32) {
        const int cur = (k0 >> 5) & 1;
        if (k0 + 32 < CDIM) {      // kick next tile into the other buffer, keep it in flight
            async_ld2(sA[cur ^ 1], gA + k0 + 32);
            async_ld2(sB[cur ^ 1], gB + k0 + 32);
            WAIT_ASYNC_PIPE();
        } else {
            WAIT_ASYNC_ALL();
        }
        __syncthreads();

        v16bf af[4], bf_[2];
#pragma unroll
        for (int mf = 0; mf < 4; ++mf) {
            const uint4* p = (const uint4*)(&As[cur][(wm + mf * 16 + ln) * 32 + (sel ? 16 : 0)]);
            af[mf] = mk16(p[0], p[1]);
        }
#pragma unroll
        for (int nf = 0; nf < 2; ++nf) {
            const uint4* p = (const uint4*)(&Bs[cur][(wn + nf * 16 + ln) * 32 + (sel ? 16 : 0)]);
            bf_[nf] = mk16(p[0], p[1]);
        }
#pragma unroll
        for (int mf = 0; mf < 4; ++mf)
#pragma unroll
            for (int nf = 0; nf < 2; ++nf)
                acc[mf][nf] = wmma_bf16(af[mf], bf_[nf], acc[mf][nf]);
        __syncthreads();
    }

    // scatter epilogue (C layout: VGPR i -> M = i + sel*8, N = ln)
#pragma unroll
    for (int mf = 0; mf < 4; ++mf) {
#pragma unroll
        for (int nf = 0; nf < 2; ++nf) {
#pragma unroll
            for (int i = 0; i < 8; ++i) {
                int m = m0 + wm + mf * 16 + i + sel * 8;
                int d = n0 + wn + nf * 16 + ln;
                float v = acc[mf][nf][i];
                int which = d >> 8, head = (d >> 5) & 7, hi = d & 31;
                int bw = m / LWIN, l = m - bw * LWIN;
                int hw = bw * NHEAD + head;
                if (which == 0)      Qs[((size_t)hw * LPAD + l) * HD + hi] = f2bf(v * SCALE);
                else if (which == 1) Kb[((size_t)hw * LPAD + l) * HD + hi] = f2bf(v);
                else                 Vtb[((size_t)hw * HD + hi) * LPAD + l] = f2bf(v);
            }
        }
    }
}

// ---------------------------------------------------------------- flash attention
// one block = one (head-window, 128-row q block); 8 waves x 16 q-rows each
__global__ void __launch_bounds__(256) attn_kernel(
    const ushort_t* __restrict__ Qs, const ushort_t* __restrict__ Kb,
    const ushort_t* __restrict__ Vtb, ushort_t* __restrict__ Ob)
{
    __shared__ ushort_t Kc[2][128 * 32];    // K chunk  [token][hd], double-buffered
    __shared__ ushort_t Vtc[2][32 * 128];   // Vt chunk [hd][token], double-buffered
    __shared__ ushort_t Pst[8 * 16 * 32];   // per-wave P staging
    const int qb = blockIdx.x;   // 0..6
    const int hw = blockIdx.y;   // 0..255
    const int bw = hw >> 3, head = hw & 7;
    const int t = threadIdx.x, lane = t & 31, wave = t >> 5;
    const int ln = lane & 15, sel = lane >> 4;
    const int q0 = qb * 128 + wave * 16;

    // Q fragment (A 16x32), one global 2xb128 pair per lane
    v16bf qa;
    {
        const uint4* p = (const uint4*)(Qs + ((size_t)hw * LPAD + q0 + ln) * HD + (sel ? 16 : 0));
        qa = mk16(p[0], p[1]);
    }

    float mst[8], lst[8];
    v8f o0 = (v8f){0,0,0,0,0,0,0,0}, o1 = (v8f){0,0,0,0,0,0,0,0};
#pragma unroll
    for (int i = 0; i < 8; ++i) { mst[i] = -1e30f; lst[i] = 0.0f; }
    ushort_t* stage = Pst + wave * 16 * 32;

    // per-thread async-copy slices (32B K + 32B Vt per chunk)
    const int krow = t >> 1, koff = (t & 1) * 16;
    const int vr = t >> 3, voff = (t & 7) * 16;
    const ushort_t* gK = Kb + ((size_t)hw * LPAD + krow) * HD + koff;   // + tb*HD per chunk
    const ushort_t* gV = Vtb + ((size_t)hw * HD + vr) * LPAD + voff;    // + tb per chunk
    const unsigned sK[2] = { lds_off(&Kc[0][krow * 32 + koff]), lds_off(&Kc[1][krow * 32 + koff]) };
    const unsigned sV[2] = { lds_off(&Vtc[0][vr * 128 + voff]), lds_off(&Vtc[1][vr * 128 + voff]) };

    // prologue: chunk 0 -> buffer 0
    async_ld2(sK[0], gK);
    async_ld2(sV[0], gV);

    for (int tb = 0; tb < LPAD; tb += 128) {
        const int cur = (tb >> 7) & 1;
        if (tb + 128 < LPAD) {   // stream next KV chunk while computing this one
            async_ld2(sK[cur ^ 1], gK + (size_t)(tb + 128) * HD);
            async_ld2(sV[cur ^ 1], gV + (tb + 128));
            WAIT_ASYNC_PIPE();
        } else {
            WAIT_ASYNC_ALL();
        }
        __syncthreads();

        for (int kv = 0; kv < 128; kv += 32) {
            // S = Q * K^T  (two 16x16 N-tiles, K-dim = hd = 32 -> one wmma each)
            v16bf kb0, kb1;
            { const uint4* p = (const uint4*)(&Kc[cur][(kv + ln) * 32 + (sel ? 16 : 0)]);      kb0 = mk16(p[0], p[1]); }
            { const uint4* p = (const uint4*)(&Kc[cur][(kv + 16 + ln) * 32 + (sel ? 16 : 0)]); kb1 = mk16(p[0], p[1]); }
            v8f z = (v8f){0,0,0,0,0,0,0,0};
            v8f s0 = wmma_bf16(qa, kb0, z);
            v8f s1 = wmma_bf16(qa, kb1, z);

            const int c0 = tb + kv + ln, c1 = c0 + 16;
#pragma unroll
            for (int i = 0; i < 8; ++i) {
                float a = (c0 < LWIN) ? s0[i] : -1e30f;   // mask padded tokens
                float b = (c1 < LWIN) ? s1[i] : -1e30f;
                float mx = fmaxf(a, b);
                mx = fmaxf(mx, __shfl_xor(mx, 1, 16));
                mx = fmaxf(mx, __shfl_xor(mx, 2, 16));
                mx = fmaxf(mx, __shfl_xor(mx, 4, 16));
                mx = fmaxf(mx, __shfl_xor(mx, 8, 16));
                float mnew = fmaxf(mst[i], mx);
                float p0 = __expf(a - mnew);
                float p1 = __expf(b - mnew);
                float corr = __expf(mst[i] - mnew);
                float rs = p0 + p1;
                rs += __shfl_xor(rs, 1, 16);
                rs += __shfl_xor(rs, 2, 16);
                rs += __shfl_xor(rs, 4, 16);
                rs += __shfl_xor(rs, 8, 16);
                lst[i] = lst[i] * corr + rs;
                mst[i] = mnew;
                o0[i] *= corr;
                o1[i] *= corr;
                int rr = i + (sel << 3);
                stage[rr * 32 + ln]      = f2bf(p0);
                stage[rr * 32 + 16 + ln] = f2bf(p1);
            }
            // P (C-layout) -> A-layout via per-wave LDS (LDS is in-order per wave)
            v16bf pa;
            { const uint4* p = (const uint4*)(stage + ln * 32 + (sel ? 16 : 0)); pa = mk16(p[0], p[1]); }
            // V fragments (B 32x16): contiguous token runs from transposed V
            v16bf vb0, vb1;
            { const uint4* p = (const uint4*)(&Vtc[cur][ln * 128 + kv + (sel ? 16 : 0)]);        vb0 = mk16(p[0], p[1]); }
            { const uint4* p = (const uint4*)(&Vtc[cur][(16 + ln) * 128 + kv + (sel ? 16 : 0)]); vb1 = mk16(p[0], p[1]); }
            o0 = wmma_bf16(pa, vb0, o0);
            o1 = wmma_bf16(pa, vb1, o1);
        }
        __syncthreads();
    }

    // normalize + store O in token-major [bw][LPAD][256] bf16
#pragma unroll
    for (int i = 0; i < 8; ++i) {
        int l = q0 + i + (sel << 3);
        if (l < LWIN) {
            float inv = 1.0f / lst[i];
            size_t base = ((size_t)bw * LPAD + l) * CDIM + head * HD;
            Ob[base + ln]      = f2bf(o0[i] * inv);
            Ob[base + 16 + ln] = f2bf(o1[i] * inv);
        }
    }
}

// ---------------------------------------------------------------- projection GEMM
__global__ void __launch_bounds__(256) proj_gemm(
    const ushort_t* __restrict__ Ob, const ushort_t* __restrict__ wproj,
    const float* __restrict__ bias, float* __restrict__ out)
{
    __shared__ ushort_t As[2][128 * 32];
    __shared__ ushort_t Bs[2][128 * 32];
    const int m0 = blockIdx.x * 128, n0 = blockIdx.y * 128;
    const int t = threadIdx.x, lane = t & 31, wave = t >> 5;
    const int ln = lane & 15, sel = lane >> 4;
    const int wm = (wave & 1) * 64, wn = (wave >> 1) * 32;

    v8f acc[4][2];
#pragma unroll
    for (int a = 0; a < 4; ++a)
#pragma unroll
        for (int b = 0; b < 2; ++b) acc[a][b] = (v8f){0,0,0,0,0,0,0,0};

    const int lrow = t >> 1, loff = (t & 1) * 16;
    const int mg = m0 + lrow;
    const int abw = mg / LWIN, al = mg - abw * LWIN;  // token -> padded O row
    const ushort_t* gA = Ob + ((size_t)abw * LPAD + al) * CDIM + loff;
    const ushort_t* gB = wproj + (size_t)(n0 + lrow) * CDIM + loff;
    const unsigned sA[2] = { lds_off(&As[0][lrow * 32 + loff]), lds_off(&As[1][lrow * 32 + loff]) };
    const unsigned sB[2] = { lds_off(&Bs[0][lrow * 32 + loff]), lds_off(&Bs[1][lrow * 32 + loff]) };

    async_ld2(sA[0], gA);
    async_ld2(sB[0], gB);

    for (int k0 = 0; k0 < CDIM; k0 += 32) {
        const int cur = (k0 >> 5) & 1;
        if (k0 + 32 < CDIM) {
            async_ld2(sA[cur ^ 1], gA + k0 + 32);
            async_ld2(sB[cur ^ 1], gB + k0 + 32);
            WAIT_ASYNC_PIPE();
        } else {
            WAIT_ASYNC_ALL();
        }
        __syncthreads();

        v16bf af[4], bf_[2];
#pragma unroll
        for (int mf = 0; mf < 4; ++mf) {
            const uint4* p = (const uint4*)(&As[cur][(wm + mf * 16 + ln) * 32 + (sel ? 16 : 0)]);
            af[mf] = mk16(p[0], p[1]);
        }
#pragma unroll
        for (int nf = 0; nf < 2; ++nf) {
            const uint4* p = (const uint4*)(&Bs[cur][(wn + nf * 16 + ln) * 32 + (sel ? 16 : 0)]);
            bf_[nf] = mk16(p[0], p[1]);
        }
#pragma unroll
        for (int mf = 0; mf < 4; ++mf)
#pragma unroll
            for (int nf = 0; nf < 2; ++nf)
                acc[mf][nf] = wmma_bf16(af[mf], bf_[nf], acc[mf][nf]);
        __syncthreads();
    }

#pragma unroll
    for (int mf = 0; mf < 4; ++mf) {
#pragma unroll
        for (int nf = 0; nf < 2; ++nf) {
#pragma unroll
            for (int i = 0; i < 8; ++i) {
                int m = m0 + wm + mf * 16 + i + sel * 8;
                int d = n0 + wn + nf * 16 + ln;
                out[(size_t)m * CDIM + d] = acc[mf][nf][i] + bias[d];
            }
        }
    }
}

// ---------------------------------------------------------------- launch
extern "C" void kernel_launch(void* const* d_in, const int* in_sizes, int n_in,
                              void* d_out, int out_size, void* d_ws, size_t ws_size,
                              hipStream_t stream) {
    const float* x      = (const float*)d_in[0];   // (2, 12544, 256)
    const float* qkv_w  = (const float*)d_in[1];   // (768, 256)
    const float* proj_w = (const float*)d_in[2];   // (256, 256)
    const float* proj_b = (const float*)d_in[3];   // (256,)

    char* ws = (char*)d_ws;
    const size_t SZ_XB = (size_t)MTOK * CDIM * 2;        // 12.85 MB
    const size_t SZ_WQ = (size_t)QKVN * CDIM * 2;
    const size_t SZ_WP = (size_t)CDIM * CDIM * 2;
    const size_t SZ_HW = (size_t)NHW * LPAD * HD * 2;    // 14.68 MB each

    ushort_t* xb    = (ushort_t*)(ws);
    ushort_t* wqkv  = (ushort_t*)(ws + SZ_XB);
    ushort_t* wproj = (ushort_t*)(ws + SZ_XB + SZ_WQ);
    ushort_t* Qs    = (ushort_t*)(ws + SZ_XB + SZ_WQ + SZ_WP);
    ushort_t* Kb    = (ushort_t*)(ws + SZ_XB + SZ_WQ + SZ_WP + SZ_HW);
    ushort_t* Vtb   = (ushort_t*)(ws + SZ_XB + SZ_WQ + SZ_WP + 2 * SZ_HW);
    ushort_t* Ob    = (ushort_t*)(ws + SZ_XB + SZ_WQ + SZ_WP + 3 * SZ_HW);

    // zero Q/K/Vt (covers the 784->896 token padding); capture-safe
    hipMemsetAsync(Qs, 0, 3 * SZ_HW, stream);

    cvt_f32_bf16<<<2048, 256, 0, stream>>>(x, xb, MTOK * CDIM);
    cvt_f32_bf16<<<192, 256, 0, stream>>>(qkv_w, wqkv, QKVN * CDIM);
    cvt_f32_bf16<<<64, 256, 0, stream>>>(proj_w, wproj, CDIM * CDIM);

    qkv_gemm<<<dim3(MTOK / 128, QKVN / 128), 256, 0, stream>>>(xb, wqkv, Qs, Kb, Vtb);
    attn_kernel<<<dim3(LPAD / 128, NHW), 256, 0, stream>>>(Qs, Kb, Vtb, Ob);
    proj_gemm<<<dim3(MTOK / 128, CDIM / 128), 256, 0, stream>>>(Ob, wproj, proj_b, (float*)d_out);
}
